// VectorQuantizer_1898375545427
// MI455X (gfx1250) — matrix operand order repaired and checked by reading, not compiled, not run on hardware
//
#include <hip/hip_runtime.h>
#include <hip/hip_bf16.h>

typedef __attribute__((ext_vector_type(16))) __bf16 v16bf;
typedef __attribute__((ext_vector_type(8)))  float  v8f;
typedef int vint4 __attribute__((vector_size(16)));   // matches async-builtin param type

#define NUM_CODES 8192
#define DIM       512
#define NTOK      32768          // 32 * 1024
#define MBLK      128            // tokens per block
#define NBLK      128            // codes per n-chunk
#define DCHUNK    32             // D-dim chunk staged in LDS (1 k-step)
#define NDCHUNK   (DIM / DCHUNK) // 16
#define DPAD      (DCHUNK + 8)   // padded LDS row (80 B, 16B-aligned)

// ---------------- CDNA5 async-copy path (probe-verified present) ----------------
#if defined(__AMDGCN__) && __has_builtin(__builtin_amdgcn_global_load_async_to_lds_b128)
#define USE_ASYNC_LDS 1
#else
#define USE_ASYNC_LDS 0
#endif

#if __has_builtin(__builtin_amdgcn_s_wait_asynccnt)
#define WAIT_ASYNC(n) __builtin_amdgcn_s_wait_asynccnt(n)
#else
#define WAIT_ASYNC(n) asm volatile("s_wait_asynccnt %0" ::"i"(n))
#endif

#if USE_ASYNC_LDS
// copy 16 bytes global -> LDS, tracked by ASYNCcnt
#define ASYNC_CP16(gptr, lptr)                                              \
    __builtin_amdgcn_global_load_async_to_lds_b128(                         \
        (__attribute__((address_space(1))) vint4*)(gptr),                   \
        (__attribute__((address_space(3))) vint4*)(lptr), 0, 0)
#endif

// ---------------- workspace layout (bytes) ----------------
static constexpr size_t CB16_OFF = 0;                                   // 8 MB
static constexpr size_t IN16_OFF = CB16_OFF + (size_t)NUM_CODES * DIM * 2;
static constexpr size_t EN_OFF   = IN16_OFF + (size_t)NTOK * DIM * 2;   // +32 MB
static constexpr size_t IDX_OFF  = EN_OFF + (size_t)NUM_CODES * 4;

// ---------------- kernel 1: codebook f32 -> bf16, ||e||^2 ----------------
__global__ void vq_prep_codebook(const float* __restrict__ cb,
                                 __bf16* __restrict__ cb16,
                                 float* __restrict__ enorm) {
    const int code = blockIdx.x;            // 8192 blocks
    const int t    = threadIdx.x;           // 128 threads, 4 elems each
    float4 v = ((const float4*)(cb + (size_t)code * DIM))[t];
    union { __bf16 b[4]; uint2 u; } pk;
    pk.b[0] = (__bf16)v.x; pk.b[1] = (__bf16)v.y;
    pk.b[2] = (__bf16)v.z; pk.b[3] = (__bf16)v.w;
    *(uint2*)(cb16 + (size_t)code * DIM + t * 4) = pk.u;
    float s = v.x*v.x + v.y*v.y + v.z*v.z + v.w*v.w;
    #pragma unroll
    for (int m = 16; m > 0; m >>= 1) s += __shfl_down(s, m, 32);
    __shared__ float red[4];
    if ((t & 31) == 0) red[t >> 5] = s;
    __syncthreads();
    if (t == 0) enorm[code] = red[0] + red[1] + red[2] + red[3];
}

// ---------------- kernel 2: inputs f32 -> bf16 (+ zero loss) ----------------
__global__ void vq_prep_inputs(const float* __restrict__ x,
                               __bf16* __restrict__ x16,
                               float* __restrict__ loss_slot, int n4) {
    const int i = blockIdx.x * blockDim.x + threadIdx.x;
    if (i == 0) *loss_slot = 0.0f;
    if (i < n4) {
        float4 v = ((const float4*)x)[i];
        union { __bf16 b[4]; uint2 u; } pk;
        pk.b[0] = (__bf16)v.x; pk.b[1] = (__bf16)v.y;
        pk.b[2] = (__bf16)v.z; pk.b[3] = (__bf16)v.w;
        ((uint2*)x16)[i] = pk.u;
    }
}

// ---------------- kernel 3: WMMA distance argmin ----------------
__global__ void __launch_bounds__(256)
vq_argmin(const __bf16* __restrict__ x16, const __bf16* __restrict__ cb16,
          const float* __restrict__ enorm,
          int* __restrict__ idx_out, float* __restrict__ idxf_out) {
    // double-buffered tiles: 2 * (128*40 + 128*40) * 2B = 40 KB
    __shared__ __align__(16) __bf16 sA[2][MBLK][DPAD];
    __shared__ __align__(16) __bf16 sB[2][NBLK][DPAD];

    const int tid   = threadIdx.x;
    const int wave  = tid >> 5;        // 0..7, one 16-row M-tile each
    const int lane  = tid & 31;
    const int lhalf = lane >> 4;       // 0 or 1
    const int l16   = lane & 15;
    const int mbase = blockIdx.x * MBLK;

    const __bf16* gA = x16 + (size_t)mbase * DIM;

    float minv[8];
    int   mini[8];
    #pragma unroll
    for (int r = 0; r < 8; ++r) { minv[r] = 3.4e38f; mini[r] = 0; }

    const v8f vzero = {};

    // staging mapping: 512 x 16B chunks per matrix per D-chunk; 2 per thread
    const int srow0 = (tid + 0)   >> 2;
    const int scol0 = ((tid + 0)  & 3) * 8;
    const int srow1 = (tid + 256) >> 2;
    const int scol1 = ((tid + 256) & 3) * 8;

    for (int nbase = 0; nbase < NUM_CODES; nbase += NBLK) {
        const __bf16* gB = cb16 + (size_t)nbase * DIM;

        v8f acc[8];
        #pragma unroll
        for (int t = 0; t < 8; ++t) acc[t] = vzero;

#if USE_ASYNC_LDS
        // ---- async double-buffered pipeline (ASYNCcnt copies) ----
        ASYNC_CP16(gA + (size_t)srow0 * DIM + scol0, &sA[0][srow0][scol0]);
        ASYNC_CP16(gB + (size_t)srow0 * DIM + scol0, &sB[0][srow0][scol0]);
        ASYNC_CP16(gA + (size_t)srow1 * DIM + scol1, &sA[0][srow1][scol1]);
        ASYNC_CP16(gB + (size_t)srow1 * DIM + scol1, &sB[0][srow1][scol1]);

        for (int d = 0; d < NDCHUNK; ++d) {
            const int cur = d & 1;
            if (d < NDCHUNK - 1) {
                const int nb = cur ^ 1;
                const int db = (d + 1) * DCHUNK;
                ASYNC_CP16(gA + (size_t)srow0 * DIM + db + scol0, &sA[nb][srow0][scol0]);
                ASYNC_CP16(gB + (size_t)srow0 * DIM + db + scol0, &sB[nb][srow0][scol0]);
                ASYNC_CP16(gA + (size_t)srow1 * DIM + db + scol1, &sA[nb][srow1][scol1]);
                ASYNC_CP16(gB + (size_t)srow1 * DIM + db + scol1, &sB[nb][srow1][scol1]);
                WAIT_ASYNC(4);   // the 4 just-issued stay outstanding; chunk d is done
            } else {
                WAIT_ASYNC(0);
            }
            __syncthreads();     // all waves' copies of chunk d landed

            // ---- compute: 8 WMMAs, B-fragment register double-buffer ----
            union { v16bf v; uint4 u[2]; } afr, bfr[2];
            const int arow = wave * 16 + l16;
            const int akb  = lhalf * 16;
            afr.u[0] = *(const uint4*)(&sA[cur][arow][akb]);
            afr.u[1] = *(const uint4*)(&sA[cur][arow][akb + 8]);
            const int bkb = lhalf * 16;
            bfr[0].u[0] = *(const uint4*)(&sB[cur][l16][bkb]);
            bfr[0].u[1] = *(const uint4*)(&sB[cur][l16][bkb + 8]);
            #pragma unroll
            for (int t = 0; t < 8; ++t) {
                if (t < 7) {
                    const int brow = (t + 1) * 16 + l16;
                    bfr[(t + 1) & 1].u[0] = *(const uint4*)(&sB[cur][brow][bkb]);
                    bfr[(t + 1) & 1].u[1] = *(const uint4*)(&sB[cur][brow][bkb + 8]);
                }
                acc[t] = __builtin_amdgcn_wmma_f32_16x16x32_bf16(
                    false, afr.v, false, bfr[t & 1].v, (short)0, acc[t], false, false);
            }
            __syncthreads();     // chunk d consumed; its buffer may be overwritten
        }
#else
        // ---- fallback: register-pipelined manual staging ----
        uint4 ra0, rb0, ra1, rb1;
        ra0 = *(const uint4*)(gA + (size_t)srow0 * DIM + scol0);
        rb0 = *(const uint4*)(gB + (size_t)srow0 * DIM + scol0);
        ra1 = *(const uint4*)(gA + (size_t)srow1 * DIM + scol1);
        rb1 = *(const uint4*)(gB + (size_t)srow1 * DIM + scol1);
        for (int d = 0; d < NDCHUNK; ++d) {
            const int cur = d & 1;
            *(uint4*)(&sA[cur][srow0][scol0]) = ra0;
            *(uint4*)(&sB[cur][srow0][scol0]) = rb0;
            *(uint4*)(&sA[cur][srow1][scol1]) = ra1;
            *(uint4*)(&sB[cur][srow1][scol1]) = rb1;
            __syncthreads();
            if (d < NDCHUNK - 1) {
                const int db = (d + 1) * DCHUNK;
                ra0 = *(const uint4*)(gA + (size_t)srow0 * DIM + db + scol0);
                rb0 = *(const uint4*)(gB + (size_t)srow0 * DIM + db + scol0);
                ra1 = *(const uint4*)(gA + (size_t)srow1 * DIM + db + scol1);
                rb1 = *(const uint4*)(gB + (size_t)srow1 * DIM + db + scol1);
            }
            union { v16bf v; uint4 u[2]; } afr, bfr[2];
            const int arow = wave * 16 + l16;
            const int akb  = lhalf * 16;
            afr.u[0] = *(const uint4*)(&sA[cur][arow][akb]);
            afr.u[1] = *(const uint4*)(&sA[cur][arow][akb + 8]);
            const int bkb = lhalf * 16;
            bfr[0].u[0] = *(const uint4*)(&sB[cur][l16][bkb]);
            bfr[0].u[1] = *(const uint4*)(&sB[cur][l16][bkb + 8]);
            #pragma unroll
            for (int t = 0; t < 8; ++t) {
                if (t < 7) {
                    const int brow = (t + 1) * 16 + l16;
                    bfr[(t + 1) & 1].u[0] = *(const uint4*)(&sB[cur][brow][bkb]);
                    bfr[(t + 1) & 1].u[1] = *(const uint4*)(&sB[cur][brow][bkb + 8]);
                }
                acc[t] = __builtin_amdgcn_wmma_f32_16x16x32_bf16(
                    false, afr.v, false, bfr[t & 1].v, (short)0, acc[t], false, false);
            }
            __syncthreads();
        }
#endif

        // distances: ||e||^2 - 2 x.e  (||x||^2 constant per row -> dropped)
        #pragma unroll
        for (int t = 0; t < 8; ++t) {
            const int code = nbase + t * 16 + l16;
            const float en = enorm[code];
            #pragma unroll
            for (int r = 0; r < 8; ++r) {
                float dd = en - 2.0f * acc[t][r];
                if (dd < minv[r]) { minv[r] = dd; mini[r] = code; }
            }
        }
    }

    // reduce across the 16 lanes of each half (C/D layout: row = r + 8*lhalf, col = l16)
    #pragma unroll
    for (int m = 1; m < 16; m <<= 1) {
        #pragma unroll
        for (int r = 0; r < 8; ++r) {
            float ov = __shfl_xor(minv[r], m, 32);
            int   oi = __shfl_xor(mini[r], m, 32);
            if (ov < minv[r] || (ov == minv[r] && oi < mini[r])) {
                minv[r] = ov; mini[r] = oi;
            }
        }
    }
    if (l16 == 0) {   // lanes 0 and 16
        #pragma unroll
        for (int r = 0; r < 8; ++r) {
            const int row = mbase + wave * 16 + r + 8 * lhalf;
            idx_out[row]  = mini[r];
            idxf_out[row] = (float)mini[r];
        }
    }
}

// ---------------- kernel 4: gather quantized rows + loss ----------------
__global__ void vq_gather_loss(const float* __restrict__ x, const float* __restrict__ cb,
                               const int* __restrict__ idx,
                               float* __restrict__ q, float* __restrict__ loss) {
    const int tok = blockIdx.x;            // 32768 blocks
    const int t   = threadIdx.x;           // 128 threads, 4 elems each
    const int code = idx[tok];
    float4 e = ((const float4*)(cb + (size_t)code * DIM))[t];
    float4 v = ((const float4*)(x  + (size_t)tok  * DIM))[t];
    ((float4*)(q + (size_t)tok * DIM))[t] = e;   // quantized_st == codebook[idx]
    float dx = e.x - v.x, dy = e.y - v.y, dz = e.z - v.z, dw = e.w - v.w;
    float s = dx*dx + dy*dy + dz*dz + dw*dw;
    #pragma unroll
    for (int m = 16; m > 0; m >>= 1) s += __shfl_down(s, m, 32);
    __shared__ float red[4];
    if ((t & 31) == 0) red[t >> 5] = s;
    __syncthreads();
    if (t == 0) {
        float tot = red[0] + red[1] + red[2] + red[3];
        // loss = q_latent + 0.25*e_latent = 1.25 * mean((q-x)^2)
        atomicAdd(loss, tot * (1.25f / ((float)NTOK * (float)DIM)));
    }
}

// ---------------- launcher ----------------
extern "C" void kernel_launch(void* const* d_in, const int* in_sizes, int n_in,
                              void* d_out, int out_size, void* d_ws, size_t ws_size,
                              hipStream_t stream) {
    const float* x  = (const float*)d_in[0];   // [32,1024,512] f32
    const float* cb = (const float*)d_in[1];   // [8192,512] f32

    float* out   = (float*)d_out;
    float* q     = out;                               // N*D floats
    float* loss  = out + (size_t)NTOK * DIM;          // 1 float
    float* idxf  = loss + 1;                          // N floats (indices)

    char*   ws    = (char*)d_ws;
    __bf16* cb16  = (__bf16*)(ws + CB16_OFF);
    __bf16* x16   = (__bf16*)(ws + IN16_OFF);
    float*  enorm = (float*)(ws + EN_OFF);
    int*    idx   = (int*)(ws + IDX_OFF);

    vq_prep_codebook<<<NUM_CODES, 128, 0, stream>>>(cb, cb16, enorm);
    const int n4 = NTOK * DIM / 4;
    vq_prep_inputs<<<(n4 + 255) / 256, 256, 0, stream>>>(x, x16, loss, n4);
    vq_argmin<<<NTOK / MBLK, 256, 0, stream>>>(x16, cb16, enorm, idx, idxf);
    vq_gather_loss<<<NTOK, 128, 0, stream>>>(x, cb, idx, q, loss);
}
